// Seq2SeqTransliterator_31112743092703
// MI455X (gfx1250) — matrix operand order repaired and checked by reading, not compiled, not run on hardware
//
#include <hip/hip_runtime.h>
#include <hip/hip_bf16.h>

// ---------------------------------------------------------------------------
// Seq2Seq LSTM (B=4096, S=T=32, E=128, H=512) on gfx1250 via bf16 WMMA.
// Weights pre-swizzled into WMMA B-fragment order (b128 loads), and the
// recurrent GEMM software-pipelines B-fragment gate-groups (double buffer)
// so loads overlap matrix ops instead of full s_wait_loadcnt drains.
// ---------------------------------------------------------------------------

typedef __attribute__((ext_vector_type(16))) __bf16 v16bf;
typedef __attribute__((ext_vector_type(8)))  __bf16 v8bf;
typedef __attribute__((ext_vector_type(8)))  float  v8f;

#define B_    4096
#define S_    32
#define E_    128
#define H_    512
#define G4H_  2048     // 4*H
#define KTOT_ 640      // E + H
#define NKT_  20       // KTOT_/32
#define NT_   128      // G4H_/16  (n-tiles of gates matrix)
#define VT_   128
#define PNT_  8        // VT_/16   (n-tiles of projection matrix)
#define PKT_  16       // H_/32
#define SOS_  1

__device__ __forceinline__ float sigmoidf_(float x) {
  return 1.0f / (1.0f + __expf(-x));
}

// Build a v16bf fragment from two contiguous 16-byte chunks.
__device__ __forceinline__ v16bf cat8(v8bf lo, v8bf hi) {
  return __builtin_shufflevector(lo, hi, 0, 1, 2, 3, 4, 5, 6, 7,
                                 8, 9, 10, 11, 12, 13, 14, 15);
}

// A fragment (16-bit A 16x32): lane half=0 holds K {0..15},
// half=1 holds K {16..31}. Two b128 loads from a bf16 row at k0.
__device__ __forceinline__ v16bf load_a32(const __bf16* __restrict__ row, int half) {
  const v8bf* q0 = (const v8bf*)(row + (half << 4));
  const v8bf* q1 = (const v8bf*)(row + 8 + (half << 4));
  return cat8(q0[0], q1[0]);
}

// B fragment from pre-swizzled weights: lane's 16 values are contiguous.
__device__ __forceinline__ v16bf load_b32(const __bf16* __restrict__ p) {
  const v8bf* q = (const v8bf*)p;
  return cat8(q[0], q[1]);
}

// ---------------------------------------------------------------------------
// One fused LSTM timestep. grid = (H/64, B/64), block = 128 (4 waves).
// Wave w: rows m0..m0+15, hidden cols hc0..hc0+63, all 4 gates.
// ---------------------------------------------------------------------------
__global__ __launch_bounds__(128)
void lstm_step_kernel(const int* __restrict__ seq,      // (B, S) tokens
                      const __bf16* __restrict__ emb,   // (V, E) bf16
                      const __bf16* __restrict__ wswz,  // swizzled (640,2048)
                      const float* __restrict__ bsum,   // (2048) bih+bhh
                      const __bf16* __restrict__ h_in,  // (B, H)
                      __bf16* __restrict__ h_out,       // (B, H)
                      float* __restrict__ cst,          // (B, H) in-place
                      const int* __restrict__ lengths,  // (B) or nullptr
                      __bf16* __restrict__ hs,          // (T, B, H) or nullptr
                      int t, int is_dec)
{
  const int tid  = threadIdx.x;
  const int wv   = tid >> 5;
  const int lane = tid & 31;
  const int ln   = lane & 15;
  const int half = lane >> 4;

  const int m0   = blockIdx.y * 64 + wv * 16;
  const int hc0  = blockIdx.x * 64;
  const int hcti = blockIdx.x * 4;     // first n-tile index of this block

  // Fused embedding gather: token for this lane's A row (row = batch index).
  const int am = m0 + ln;
  int tok;
  if (!is_dec) tok = seq[am * S_ + t];
  else         tok = (t == 0) ? SOS_ : seq[am * S_ + t - 1];
  const __bf16* xrow = emb  + (size_t)tok * E_;
  const __bf16* hrow = h_in + (size_t)am  * H_;

  const v8f vzero = {0.f, 0.f, 0.f, 0.f, 0.f, 0.f, 0.f, 0.f};
  v8f acc[4][4];
#pragma unroll
  for (int g = 0; g < 4; ++g)
#pragma unroll
    for (int nt = 0; nt < 4; ++nt)
      acc[g][nt] = vzero;

  // ---- software-pipelined K loop -----------------------------------------
  v16bf a  = load_a32(xrow, half);     // kt = 0 A fragment (K 0..31 from x)
  v16bf an = a;
  v16bf bb[2][4];                      // double-buffered gate-group of 4 B frags
  {
    const __bf16* p0 = wswz + (size_t)hcti * 512 + lane * 16;
#pragma unroll
    for (int nt = 0; nt < 4; ++nt) bb[0][nt] = load_b32(p0 + (size_t)nt * 512);
  }

#pragma unroll 1
  for (int kt = 0; kt < NKT_; ++kt) {
#pragma unroll
    for (int g = 0; g < 4; ++g) {
      const int cur = g & 1;
      const int nxt = cur ^ 1;
      // Issue loads for the NEXT gate-group (possibly in the next K-tile)
      const int nkt = (g == 3) ? kt + 1 : kt;
      const int ng  = (g == 3) ? 0 : g + 1;
      if (nkt < NKT_) {
        const __bf16* p = wswz + ((size_t)nkt * NT_ + hcti) * 512
                               + (size_t)(ng * 32) * 512 + lane * 16;
#pragma unroll
        for (int nt = 0; nt < 4; ++nt)
          bb[nxt][nt] = load_b32(p + (size_t)nt * 512);
      }
      // Prefetch next A fragment while group 0 computes
      if (g == 0 && kt + 1 < NKT_) {
        const int k0n = (kt + 1) * 32;
        an = load_a32((k0n < E_) ? (xrow + k0n) : (hrow + (k0n - E_)), half);
      }
      // Prefetch the K-tile after next into cache
      if (g == 2 && kt + 2 < NKT_)
        __builtin_prefetch(wswz + ((size_t)(kt + 2) * NT_ + hcti) * 512
                               + lane * 16, 0, 3);
      // 4 WMMAs of the current gate-group
#pragma unroll
      for (int nt = 0; nt < 4; ++nt)
        acc[g][nt] = __builtin_amdgcn_wmma_f32_16x16x32_bf16(
            false, a, false, bb[cur][nt], (short)0, acc[g][nt], false, false);
    }
    a = an;
  }

  // ---- pointwise LSTM epilogue (C layout: lane -> col n, VGPR r -> row) ----
#pragma unroll
  for (int nt = 0; nt < 4; ++nt) {
    const int hcc = hc0 + nt * 16 + ln;
    const float bi = bsum[hcc];
    const float bf = bsum[H_ + hcc];
    const float bg = bsum[2 * H_ + hcc];
    const float bo = bsum[3 * H_ + hcc];
#pragma unroll
    for (int r = 0; r < 8; ++r) {
      const int mm = m0 + half * 8 + r;
      const size_t idx = (size_t)mm * H_ + hcc;
      float gi = sigmoidf_(acc[0][nt][r] + bi);
      float gf = sigmoidf_(acc[1][nt][r] + bf);
      float gg = tanhf(acc[2][nt][r] + bg);
      float go = sigmoidf_(acc[3][nt][r] + bo);
      float cold = cst[idx];
      float cnew = gf * cold + gi * gg;
      float hnew = go * tanhf(cnew);
      if (lengths && t >= lengths[mm]) {   // packed-sequence mask (encoder)
        cnew = cold;
        hnew = (float)h_in[idx];
      }
      cst[idx]   = cnew;
      h_out[idx] = (__bf16)hnew;
      if (hs) hs[((size_t)t * B_ + mm) * H_ + hcc] = (__bf16)hnew;
    }
  }
}

// ---------------------------------------------------------------------------
// Output projection: logits[b,t,v] = hs[t,b,:] . W_out[v,:] + b_out[v]
// M = T*B (row m = t*B + b), K = H, N = VT. grid = (VT/64, T*B/64), block 128.
// ---------------------------------------------------------------------------
__global__ __launch_bounds__(128)
void proj_kernel(const __bf16* __restrict__ hs,    // (T*B, H)
                 const __bf16* __restrict__ wswz,  // swizzled (H, VT)
                 const float* __restrict__ bout,   // (VT)
                 float* __restrict__ out)          // (B, T, VT) fp32
{
  const int tid  = threadIdx.x;
  const int wv   = tid >> 5;
  const int lane = tid & 31;
  const int ln   = lane & 15;
  const int half = lane >> 4;

  const int m0  = blockIdx.y * 64 + wv * 16;
  const int n0  = blockIdx.x * 64;
  const int nti = blockIdx.x * 4;

  const int am = m0 + ln;
  const __bf16* ar = hs + (size_t)am * H_;

  const v8f vzero = {0.f, 0.f, 0.f, 0.f, 0.f, 0.f, 0.f, 0.f};
  v8f acc[4];
#pragma unroll
  for (int nt = 0; nt < 4; ++nt) acc[nt] = vzero;

#pragma unroll 1
  for (int kt = 0; kt < PKT_; ++kt) {
    v16bf a = load_a32(ar + kt * 32, half);
    const __bf16* wk = wswz + ((size_t)kt * PNT_ + nti) * 512 + lane * 16;
#pragma unroll
    for (int nt = 0; nt < 4; ++nt) {
      v16bf b = load_b32(wk + (size_t)nt * 512);
      acc[nt] = __builtin_amdgcn_wmma_f32_16x16x32_bf16(
          false, a, false, b, (short)0, acc[nt], false, false);
    }
  }

#pragma unroll
  for (int nt = 0; nt < 4; ++nt) {
    const int v = n0 + nt * 16 + ln;
    const float bias = bout[v];
#pragma unroll
    for (int r = 0; r < 8; ++r) {
      const int mm = m0 + half * 8 + r;
      const int tt = mm >> 12;          // mm / B_
      const int bb = mm & (B_ - 1);     // mm % B_
      out[((size_t)bb * S_ + tt) * VT_ + v] = acc[nt][r] + bias;
    }
  }
}

// ---------------------------------------------------------------------------
// Weight swizzle: fragment tile (kt, ntile) holds 32 lanes x 16 bf16 where
// lane L, elem j  <-  src[n][k], n = ntile*16 + L%16,
//                                k = kt*32 + (L/16)*16 + j   (k local to src).
// src is (Nsrc, Ksrc) row-major fp32, placed at dst K-offset koff.
// dst tile stride = 512 elements; NT = number of n-tiles in dst.
// ---------------------------------------------------------------------------
__global__ void swizzle_weights_kernel(__bf16* __restrict__ dst,
                                       const float* __restrict__ src,
                                       int Ksrc, int koff, int NT) {
  int i = blockIdx.x * blockDim.x + threadIdx.x;
  int total = (Ksrc >> 5) * NT * 512;
  if (i >= total) return;
  int j   = i & 15;
  int L   = (i >> 4) & 31;
  int nt  = (i >> 9) % NT;
  int ktl = (i >> 9) / NT;
  int n   = nt * 16 + (L & 15);
  int k   = ktl * 32 + ((L >> 4) << 4) + j;
  int ktg = (koff >> 5) + ktl;
  dst[(((size_t)ktg * NT + nt) << 9) + (L << 4) + j] =
      (__bf16)src[(size_t)n * Ksrc + k];
}

__global__ void f32_to_bf16_kernel(__bf16* __restrict__ dst,
                                   const float* __restrict__ src, int n) {
  int i = blockIdx.x * blockDim.x + threadIdx.x;
  if (i < n) dst[i] = (__bf16)src[i];
}

__global__ void bias_sum_kernel(float* __restrict__ dst,
                                const float* __restrict__ a,
                                const float* __restrict__ b, int n) {
  int i = blockIdx.x * blockDim.x + threadIdx.x;
  if (i < n) dst[i] = a[i] + b[i];
}

// ---------------------------------------------------------------------------
extern "C" void kernel_launch(void* const* d_in, const int* in_sizes, int n_in,
                              void* d_out, int out_size, void* d_ws, size_t ws_size,
                              hipStream_t stream) {
  (void)in_sizes; (void)n_in; (void)out_size; (void)ws_size;

  const int*   src_seq = (const int*)  d_in[0];
  const int*   src_len = (const int*)  d_in[1];
  const int*   tgt_seq = (const int*)  d_in[2];
  const float* emb_src = (const float*)d_in[3];
  const float* enc_Wih = (const float*)d_in[4];
  const float* enc_Whh = (const float*)d_in[5];
  const float* enc_bih = (const float*)d_in[6];
  const float* enc_bhh = (const float*)d_in[7];
  const float* emb_tgt = (const float*)d_in[8];
  const float* dec_Wih = (const float*)d_in[9];
  const float* dec_Whh = (const float*)d_in[10];
  const float* dec_bih = (const float*)d_in[11];
  const float* dec_bhh = (const float*)d_in[12];
  const float* W_out   = (const float*)d_in[13];
  const float* b_out   = (const float*)d_in[14];
  float* out = (float*)d_out;

  size_t off = 0;
  auto take = [&](size_t bytes) -> char* {
    char* p = (char*)d_ws + off;
    off += (bytes + 255) & ~(size_t)255;
    return p;
  };
  __bf16* wswz_e  = (__bf16*)take((size_t)NKT_ * NT_ * 512 * 2);  // 2.5 MB
  __bf16* wswz_d  = (__bf16*)take((size_t)NKT_ * NT_ * 512 * 2);  // 2.5 MB
  __bf16* woswz   = (__bf16*)take((size_t)PKT_ * PNT_ * 512 * 2); // 128 KB
  __bf16* embs_bf = (__bf16*)take((size_t)96 * E_ * 2);
  __bf16* embt_bf = (__bf16*)take((size_t)VT_ * E_ * 2);
  float*  bsum_e  = (float*) take((size_t)G4H_ * 4);
  float*  bsum_d  = (float*) take((size_t)G4H_ * 4);
  __bf16* hA      = (__bf16*)take((size_t)B_ * H_ * 2);           // 4 MB
  __bf16* hB      = (__bf16*)take((size_t)B_ * H_ * 2);           // 4 MB
  float*  cbuf    = (float*) take((size_t)B_ * H_ * 4);           // 8 MB
  __bf16* hsbuf   = (__bf16*)take((size_t)S_ * B_ * H_ * 2);      // 128 MB

  // h0 = 0 (bf16 zero is all-zero bits), c0 = 0
  hipMemsetAsync(hA,   0, (size_t)B_ * H_ * 2, stream);
  hipMemsetAsync(cbuf, 0, (size_t)B_ * H_ * 4, stream);

  // Swizzle Wih (K 0..127) and Whh (K 128..639) into combined fragment order.
  {
    int tot = (E_ >> 5) * NT_ * 512;   // 262144
    swizzle_weights_kernel<<<(tot + 255) / 256, 256, 0, stream>>>(
        wswz_e, enc_Wih, E_, 0, NT_);
    swizzle_weights_kernel<<<(tot + 255) / 256, 256, 0, stream>>>(
        wswz_d, dec_Wih, E_, 0, NT_);
  }
  {
    int tot = (H_ >> 5) * NT_ * 512;   // 1048576
    swizzle_weights_kernel<<<(tot + 255) / 256, 256, 0, stream>>>(
        wswz_e, enc_Whh, H_, E_, NT_);
    swizzle_weights_kernel<<<(tot + 255) / 256, 256, 0, stream>>>(
        wswz_d, dec_Whh, H_, E_, NT_);
  }
  {
    int tot = (H_ >> 5) * PNT_ * 512;  // 65536
    swizzle_weights_kernel<<<(tot + 255) / 256, 256, 0, stream>>>(
        woswz, W_out, H_, 0, PNT_);
  }
  f32_to_bf16_kernel<<<(96 * E_ + 255) / 256, 256, 0, stream>>>(
      embs_bf, emb_src, 96 * E_);
  f32_to_bf16_kernel<<<(VT_ * E_ + 255) / 256, 256, 0, stream>>>(
      embt_bf, emb_tgt, VT_ * E_);
  bias_sum_kernel<<<(G4H_ + 255) / 256, 256, 0, stream>>>(bsum_e, enc_bih, enc_bhh, G4H_);
  bias_sum_kernel<<<(G4H_ + 255) / 256, 256, 0, stream>>>(bsum_d, dec_bih, dec_bhh, G4H_);

  const dim3 sgrid(H_ / 64, B_ / 64);   // (8, 64)

  // Encoder: 32 masked steps, ping-pong h between hA/hB.
  __bf16* hin = hA;
  __bf16* hout = hB;
  for (int t = 0; t < S_; ++t) {
    lstm_step_kernel<<<sgrid, 128, 0, stream>>>(
        src_seq, embs_bf, wswz_e, bsum_e, hin, hout, cbuf,
        src_len, (__bf16*)nullptr, t, /*is_dec=*/0);
    __bf16* tmp = hin; hin = hout; hout = tmp;
  }

  // Decoder: 32 teacher-forced steps, store h into hsbuf.
  for (int t = 0; t < S_; ++t) {
    lstm_step_kernel<<<sgrid, 128, 0, stream>>>(
        tgt_seq, embt_bf, wswz_d, bsum_d, hin, hout, cbuf,
        (const int*)nullptr, hsbuf, t, /*is_dec=*/1);
    __bf16* tmp = hin; hin = hout; hout = tmp;
  }

  // Logits projection.
  const dim3 pgrid(VT_ / 64, (S_ * B_) / 64);  // (2, 2048)
  proj_kernel<<<pgrid, 128, 0, stream>>>(hsbuf, woswz, b_out, out);
}
